// GPT2LMHeadModel_49907519980143
// MI455X (gfx1250) — matrix operand order repaired and checked by reading, not compiled, not run on hardware
//
#include <hip/hip_runtime.h>
#include <hip/hip_bf16.h>
#include <math.h>

// ---------------- model constants ----------------
#define Vv   50257
#define E    768
#define Hh   12
#define Ll   4
#define Ss   1024
#define HID  3072
#define Bb   2
#define HE   (Hh * E)          // 9216
#define BS   (Bb * Ss)         // 2048
#define EPS  1e-5f

typedef __bf16 bf16;
typedef __attribute__((ext_vector_type(16))) __bf16 v16bf;
typedef __attribute__((ext_vector_type(8)))  float  v8f;

// async-to-LDS builtin operand types: 16B int vector in global (as1) / LDS (as3)
typedef int v4i __attribute__((vector_size(16)));
typedef __attribute__((address_space(1))) v4i gv4i;
typedef __attribute__((address_space(3))) v4i lv4i;

// 16-byte memory->LDS copy. Prefer CDNA5 async-to-LDS (ASYNCcnt-tracked,
// no VGPR round trip); fall back to sync load+store if builtin absent.
__device__ __forceinline__ void async_copy16(const bf16* g, bf16* l) {
#if __has_builtin(__builtin_amdgcn_global_load_async_to_lds_b128)
    __builtin_amdgcn_global_load_async_to_lds_b128(
        (gv4i*)(unsigned long long)g,
        (lv4i*)(unsigned int)(unsigned long long)l,
        0, 0);
#else
    *(uint4*)l = *(const uint4*)g;
#endif
}

__device__ __forceinline__ void wait_async0() {
#if __has_builtin(__builtin_amdgcn_s_wait_asynccnt)
    __builtin_amdgcn_s_wait_asynccnt(0);
#else
    asm volatile("s_wait_asynccnt 0x0" ::: "memory");
#endif
}

// ---------------- GEMM: C = A(MxK) * B or B^T, bf16 in, f32 acc via WMMA ----
// A row-major [M,K] bf16.
// TRANSB=0: B row-major [K,N]. TRANSB=1: B row-major [N,K] (compute A*B^T).
// OUTBF16: C stored as bf16. GELU_: exact gelu epilogue.
// Batched via gridDim.z: z -> (zb=z/zdiv, zh=z%zdiv), offsets zb*s?b + zh*s?h.
#define BM 128
#define BN 128
#define BK 32
#define LDSS 40   // padded bf16 row stride (80B rows; 16B-aligned chunks)

union FragU { v16bf v; uint4 q[2]; };

template<bool TRANSB, bool OUTBF16, bool GELU_>
__global__ __launch_bounds__(256)
void gemm_bf16_wmma(const bf16* __restrict__ A, long lda,
                    const bf16* __restrict__ Bm, long ldb,
                    void* __restrict__ Cv, long ldc,
                    const float* __restrict__ bias,
                    int M, int N, int K, float alpha,
                    int zdiv, long sAb, long sAh, long sBb, long sBh,
                    long sCb, long sCh)
{
    __shared__ bf16 As[2][BM * LDSS];
    __shared__ bf16 Bs[2][BN * LDSS];   // N-major: Bs[n][k]

    const int z  = blockIdx.z;
    const int zb = z / zdiv, zh = z % zdiv;
    const bf16* Ab  = A  + zb * sAb + zh * sAh;
    const bf16* Bb_ = Bm + zb * sBb + zh * sBh;
    const long  cz  = zb * sCb + zh * sCh;

    const int tid  = threadIdx.x;
    const int wave = tid >> 5;
    const int lane = tid & 31;
    const int wm   = wave & 3;     // 4 waves along M
    const int wn   = wave >> 2;    // 2 waves along N
    const int m0   = blockIdx.y * BM;
    const int n0   = blockIdx.x * BN;
    const int lr    = lane & 15;
    const int lhalf = lane >> 4;

    // staging coordinates (2 chunks of 8 bf16 per thread, unconditional)
    const int cA0 = tid,        cA1 = tid + 256;
    const int rA0 = cA0 >> 2,   kA0 = (cA0 & 3) << 3;   // BK/8 = 4 chunks/row
    const int rA1 = cA1 >> 2,   kA1 = (cA1 & 3) << 3;

    v8f acc[2][4];
#pragma unroll
    for (int i = 0; i < 2; ++i)
#pragma unroll
        for (int j = 0; j < 4; ++j) acc[i][j] = v8f{};

    auto stage = [&](int k0, int buf) {
        // ---- A tile (BM x BK): async memory->LDS ----
        async_copy16(Ab + (long)(m0 + rA0) * lda + k0 + kA0, &As[buf][rA0 * LDSS + kA0]);
        async_copy16(Ab + (long)(m0 + rA1) * lda + k0 + kA1, &As[buf][rA1 * LDSS + kA1]);
        if (TRANSB) {
            // global [N,K]: contiguous along k -> async straight into Bs[n][k]
            async_copy16(Bb_ + (long)(n0 + rA0) * ldb + k0 + kA0, &Bs[buf][rA0 * LDSS + kA0]);
            async_copy16(Bb_ + (long)(n0 + rA1) * ldb + k0 + kA1, &Bs[buf][rA1 * LDSS + kA1]);
        } else {
            // global [K,N]: contiguous along n; load both chunks, then
            // scatter-transpose into Bs[n][k] with 16-bit DS stores.
            const int k0c = cA0 >> 4, n0c = (cA0 & 15) << 3;  // BN/8 = 16 chunks/k-row
            const int k1c = cA1 >> 4, n1c = (cA1 & 15) << 3;
            uint4 d0 = *(const uint4*)(Bb_ + (long)(k0 + k0c) * ldb + n0 + n0c);
            uint4 d1 = *(const uint4*)(Bb_ + (long)(k0 + k1c) * ldb + n0 + n1c);
            const bf16* p0 = (const bf16*)&d0;
            const bf16* p1 = (const bf16*)&d1;
#pragma unroll
            for (int j = 0; j < 8; ++j)
                Bs[buf][(n0c + j) * LDSS + k0c] = p0[j];
#pragma unroll
            for (int j = 0; j < 8; ++j)
                Bs[buf][(n1c + j) * LDSS + k1c] = p1[j];
        }
    };

    const int nt = K / BK;
    stage(0, 0);   // prologue

    for (int t = 0; t < nt; ++t) {
        const int cur = t & 1;
        wait_async0();        // my async copies for tile t are in LDS
        __syncthreads();      // everyone's copies visible; prev buffer free
        if (t + 1 < nt) stage((t + 1) * BK, cur ^ 1);   // overlap with compute

        // ---- fragments per CDNA5 16-bit WMMA lane layout ----
        FragU afrag[2], bfrag[4];
        const int kbA = lhalf * 8;
#pragma unroll
        for (int i = 0; i < 2; ++i) {
            const bf16* p = &As[cur][(wm * 32 + i * 16 + lr) * LDSS];
            afrag[i].q[0] = *(const uint4*)(p + kbA);
            afrag[i].q[1] = *(const uint4*)(p + kbA + 16);
        }
        const int kbB = lhalf * 8;
#pragma unroll
        for (int j = 0; j < 4; ++j) {
            const bf16* p = &Bs[cur][(wn * 64 + j * 16 + lr) * LDSS + kbB];
            bfrag[j].q[0] = *(const uint4*)(p);
            bfrag[j].q[1] = *(const uint4*)(p + 16);
        }
#pragma unroll
        for (int i = 0; i < 2; ++i)
#pragma unroll
            for (int j = 0; j < 4; ++j)
                acc[i][j] = __builtin_amdgcn_wmma_f32_16x16x32_bf16(
                    false, afrag[i].v, false, bfrag[j].v,
                    (short)0, acc[i][j], false, false);
    }

    // ---- epilogue: D f32 16x16 layout (lane col = lr, rows v + lhalf*8) ----
#pragma unroll
    for (int i = 0; i < 2; ++i) {
        const int rbase = m0 + wm * 32 + i * 16 + lhalf * 8;
#pragma unroll
        for (int j = 0; j < 4; ++j) {
            const int col = n0 + wn * 64 + j * 16 + lr;
            const float bvv = bias ? bias[col] : 0.f;
#pragma unroll
            for (int v = 0; v < 8; ++v) {
                float val = acc[i][j][v] * alpha + bvv;
                if (GELU_) val = 0.5f * val * (1.f + erff(val * 0.70710678f));
                const long idx = cz + (long)(rbase + v) * ldc + col;
                if (OUTBF16) ((bf16*)Cv)[idx] = (bf16)val;
                else         ((float*)Cv)[idx] = val;
            }
        }
    }
}

// ---------------- elementwise / reduction kernels ----------------
__device__ inline float block_sum(float v, float* red) {
    const int tid = threadIdx.x;
    red[tid] = v; __syncthreads();
    for (int s = 128; s > 0; s >>= 1) {
        if (tid < s) red[tid] += red[tid + s];
        __syncthreads();
    }
    const float r = red[0]; __syncthreads();
    return r;
}
__device__ inline float block_max(float v, float* red) {
    const int tid = threadIdx.x;
    red[tid] = v; __syncthreads();
    for (int s = 128; s > 0; s >>= 1) {
        if (tid < s) red[tid] = fmaxf(red[tid], red[tid + s]);
        __syncthreads();
    }
    const float r = red[0]; __syncthreads();
    return r;
}

__global__ void embed_kernel(const float* __restrict__ tok,
                             const float* __restrict__ pos,
                             const int* __restrict__ x,
                             float* __restrict__ h, bf16* __restrict__ hb) {
    const long i = (long)blockIdx.x * 256 + threadIdx.x;   // B*S*E total
    const int  e  = (int)(i % E);
    const long bs = i / E;
    const int  s  = (int)(bs % Ss);
    const float v = tok[(long)x[bs] * E + e] + pos[(long)s * E + e];
    h[i] = v; hb[i] = (bf16)v;
}

__global__ void f32_to_bf16_kernel(const float* __restrict__ in,
                                   bf16* __restrict__ out, long n) {
    const long i = (long)blockIdx.x * 256 + threadIdx.x;
    if (i < n) out[i] = (bf16)in[i];
}

// causal softmax, in-place on bf16 scores; one wave32 per row
__global__ void softmax_causal(bf16* __restrict__ sc) {
    const long row = blockIdx.x;            // z*S + q
    const int  q   = (int)(row % Ss);
    bf16* p = sc + row * Ss;
    const int lane = threadIdx.x;
    float vals[Ss / 32];
    float mx = -INFINITY;
#pragma unroll
    for (int i = 0; i < Ss / 32; ++i) {
        const int k = i * 32 + lane;
        const float v = (k <= q) ? (float)p[k] : -INFINITY;
        vals[i] = v; mx = fmaxf(mx, v);
    }
#pragma unroll
    for (int m = 16; m >= 1; m >>= 1) mx = fmaxf(mx, __shfl_xor(mx, m, 32));
    float sum = 0.f;
#pragma unroll
    for (int i = 0; i < Ss / 32; ++i) {
        const float e = (vals[i] == -INFINITY) ? 0.f : __expf(vals[i] - mx);
        vals[i] = e; sum += e;
    }
#pragma unroll
    for (int m = 16; m >= 1; m >>= 1) sum += __shfl_xor(sum, m, 32);
    const float inv = 1.f / sum;
#pragma unroll
    for (int i = 0; i < Ss / 32; ++i) p[i * 32 + lane] = (bf16)(vals[i] * inv);
}

// y = LN(a + r)*g + bt ; writes f32 out and optional bf16 out
__global__ __launch_bounds__(256)
void ln_kernel(const float* __restrict__ a, const float* __restrict__ r,
               const float* __restrict__ g, const float* __restrict__ bt,
               float* __restrict__ out, bf16* __restrict__ outb) {
    __shared__ float red[256];
    const long base = (long)blockIdx.x * E;
    const int tid = threadIdx.x;
    float x[3]; float sum = 0.f;
#pragma unroll
    for (int i = 0; i < 3; ++i) {
        const int e = tid + i * 256;
        const float v = a[base + e] + r[base + e];
        x[i] = v; sum += v;
    }
    sum = block_sum(sum, red);
    const float m = sum * (1.f / E);
    float sq = 0.f;
#pragma unroll
    for (int i = 0; i < 3; ++i) { const float d = x[i] - m; sq += d * d; }
    sq = block_sum(sq, red);
    const float inv = rsqrtf(sq * (1.f / E) + EPS);
#pragma unroll
    for (int i = 0; i < 3; ++i) {
        const int e = tid + i * 256;
        const float y = (x[i] - m) * inv * g[e] + bt[e];
        out[base + e] = y;
        if (outb) outb[base + e] = (bf16)y;
    }
}

// BatchNorm1d over (batch, embed) per sequence position s; in-place on h
__global__ __launch_bounds__(256)
void bn_kernel(float* __restrict__ h, bf16* __restrict__ hb,
               const float* __restrict__ g, const float* __restrict__ bt) {
    __shared__ float red[256];
    const int s = blockIdx.x;
    const int tid = threadIdx.x;
    float x[6]; float sum = 0.f, sq = 0.f;
#pragma unroll
    for (int i = 0; i < 6; ++i) {
        const int t = tid + i * 256;         // 0..1535 = (b,e)
        const int b = t / E, e = t % E;
        const float v = h[((long)b * Ss + s) * E + e];
        x[i] = v; sum += v; sq += v * v;
    }
    sum = block_sum(sum, red);
    sq  = block_sum(sq, red);
    const float m   = sum * (1.f / (Bb * E));
    const float var = sq * (1.f / (Bb * E)) - m * m;
    const float sc  = g[s] * rsqrtf(var + EPS);
    const float sh  = bt[s] - m * sc;
#pragma unroll
    for (int i = 0; i < 6; ++i) {
        const int t = tid + i * 256;
        const int b = t / E, e = t % E;
        const long idx = ((long)b * Ss + s) * E + e;
        const float y = x[i] * sc + sh;
        h[idx] = y; hb[idx] = (bf16)y;
    }
}

__global__ void pool_kernel(const float* __restrict__ h, float* __restrict__ pooled) {
    const int i = blockIdx.x * 256 + threadIdx.x;   // B*E
    if (i >= Bb * E) return;
    const int b = i / E, e = i % E;
    float s = 0.f;
    for (int t = 0; t < Ss; ++t) s += h[((long)b * Ss + t) * E + e];
    pooled[i] = s * (1.f / Ss);
}

__global__ void lmhead_kernel(const float* __restrict__ pooled,
                              const float* __restrict__ lmW,
                              const float* __restrict__ lmb,
                              float* __restrict__ logits) {
    const int v = blockIdx.x * 256 + threadIdx.x;
    const int b = blockIdx.y;
    if (v >= Vv) return;
    float s = lmb[v];
    for (int e = 0; e < E; ++e) s += pooled[b * E + e] * lmW[(long)e * Vv + v];
    logits[(long)b * Vv + v] = s;
}

__global__ __launch_bounds__(256)
void logsoftmax_kernel(const float* __restrict__ logits, float* __restrict__ out) {
    __shared__ float red[256];
    const int b = blockIdx.x;
    const int tid = threadIdx.x;
    const float* p = logits + (long)b * Vv;
    float mx = -INFINITY;
    for (int v = tid; v < Vv; v += 256) mx = fmaxf(mx, p[v]);
    mx = block_max(mx, red);
    float sum = 0.f;
    for (int v = tid; v < Vv; v += 256) sum += __expf(p[v] - mx);
    sum = block_sum(sum, red);
    const float lse = mx + logf(sum);
    for (int v = tid; v < Vv; v += 256) out[(long)b * Vv + v] = p[v] - lse;
}

// ---------------- host orchestration ----------------
extern "C" void kernel_launch(void* const* d_in, const int* in_sizes, int n_in,
                              void* d_out, int out_size, void* d_ws, size_t ws_size,
                              hipStream_t stream) {
    (void)in_sizes; (void)n_in; (void)out_size; (void)ws_size;
    const float* tok_emb = (const float*)d_in[0];
    const float* pos_emb = (const float*)d_in[1];
    const float* Wq  = (const float*)d_in[2];
    const float* Wk  = (const float*)d_in[3];
    const float* Wv  = (const float*)d_in[4];
    const float* Wo  = (const float*)d_in[5];
    const float* bo  = (const float*)d_in[6];
    const float* ln1g = (const float*)d_in[7];
    const float* ln1b = (const float*)d_in[8];
    const float* ln2g = (const float*)d_in[9];
    const float* ln2b = (const float*)d_in[10];
    const float* W1  = (const float*)d_in[11];
    const float* b1  = (const float*)d_in[12];
    const float* W2  = (const float*)d_in[13];
    const float* b2  = (const float*)d_in[14];
    const float* bng = (const float*)d_in[15];
    const float* bnb = (const float*)d_in[16];
    const float* lmW = (const float*)d_in[17];
    const float* lmb = (const float*)d_in[18];
    const int*   x   = (const int*)d_in[19];
    float* outp = (float*)d_out;

    // scratch carve-out
    char* w = (char*)d_ws;
    auto alloc = [&](size_t bytes) -> void* {
        void* p = (void*)w;
        w += (bytes + 255) & ~(size_t)255;
        return p;
    };
    float* hA    = (float*)alloc((size_t)BS * E * 4);
    float* hB    = (float*)alloc((size_t)BS * E * 4);
    float* P     = (float*)alloc((size_t)BS * E * 4);
    bf16*  hb    = (bf16*) alloc((size_t)BS * E * 2);
    bf16*  wbuf  = (bf16*) alloc((size_t)E * HE * 2);   // biggest weight
    bf16*  Qb    = (bf16*) alloc((size_t)BS * HE * 2);
    bf16*  Kb    = (bf16*) alloc((size_t)BS * HE * 2);
    bf16*  Vb    = (bf16*) alloc((size_t)BS * HE * 2);
    bf16*  Sc    = (bf16*) alloc((size_t)Bb * Hh * Ss * Ss * 2);
    bf16*  attnb = (bf16*) alloc((size_t)BS * HE * 2);
    bf16*  ff1b  = (bf16*) alloc((size_t)BS * HID * 2);
    float* pooled= (float*)alloc((size_t)Bb * E * 4);
    float* logits= (float*)alloc((size_t)Bb * Vv * 4);

    const float scale = 0.03608439182435161f;   // 1/sqrt(768)

    // embedding
    embed_kernel<<<(BS * E) / 256, 256, 0, stream>>>(tok_emb, pos_emb, x, hA, hb);

    for (int l = 0; l < Ll; ++l) {
        const long EHE = (long)E * HE;
        // ---- QKV projections (NN, out bf16) ----
        const float* Wqkv[3] = { Wq + l * EHE, Wk + l * EHE, Wv + l * EHE };
        bf16* Out3[3] = { Qb, Kb, Vb };
        for (int t = 0; t < 3; ++t) {
            f32_to_bf16_kernel<<<(int)((EHE + 255) / 256), 256, 0, stream>>>(Wqkv[t], wbuf, EHE);
            dim3 g(HE / BN, BS / BM, 1);
            gemm_bf16_wmma<false, true, false><<<g, 256, 0, stream>>>(
                hb, E, wbuf, HE, Out3[t], HE, nullptr,
                BS, HE, E, 1.f, 1, 0, 0, 0, 0, 0, 0);
        }
        // ---- scores = Q K^T * scale (NT, batched over B*H, out bf16) ----
        {
            dim3 g(Ss / BN, Ss / BM, Bb * Hh);
            gemm_bf16_wmma<true, true, false><<<g, 256, 0, stream>>>(
                Qb, HE, Kb, HE, Sc, Ss, nullptr,
                Ss, Ss, E, scale, Hh,
                (long)Ss * HE, (long)E, (long)Ss * HE, (long)E,
                (long)Hh * Ss * Ss, (long)Ss * Ss);
        }
        // ---- causal softmax in place ----
        softmax_causal<<<Bb * Hh * Ss, 32, 0, stream>>>(Sc);
        // ---- attn = probs @ V (NN, batched, out bf16) ----
        {
            dim3 g(E / BN, Ss / BM, Bb * Hh);
            gemm_bf16_wmma<false, true, false><<<g, 256, 0, stream>>>(
                Sc, Ss, Vb, HE, attnb, HE, nullptr,
                Ss, E, Ss, 1.f, Hh,
                (long)Hh * Ss * Ss, (long)Ss * Ss,
                (long)Ss * HE, (long)E, (long)Ss * HE, (long)E);
        }
        // ---- attn @ Wo + bo -> P (f32) ----
        {
            f32_to_bf16_kernel<<<(int)((EHE + 255) / 256), 256, 0, stream>>>(Wo + l * EHE, wbuf, EHE);
            dim3 g(E / BN, BS / BM, 1);
            gemm_bf16_wmma<false, false, false><<<g, 256, 0, stream>>>(
                attnb, HE, wbuf, E, P, E, bo + l * E,
                BS, E, HE, 1.f, 1, 0, 0, 0, 0, 0, 0);
        }
        // ---- h = LN1(P + hA) -> hB (f32) + hb (bf16) ----
        ln_kernel<<<BS, 256, 0, stream>>>(P, hA, ln1g + l * E, ln1b + l * E, hB, hb);
        // ---- FFN1: gelu(h @ W1 + b1) -> ff1b (bf16) ----
        {
            const long EH1 = (long)E * HID;
            f32_to_bf16_kernel<<<(int)((EH1 + 255) / 256), 256, 0, stream>>>(W1 + l * EH1, wbuf, EH1);
            dim3 g(HID / BN, BS / BM, 1);
            gemm_bf16_wmma<false, true, true><<<g, 256, 0, stream>>>(
                hb, E, wbuf, HID, ff1b, HID, b1 + l * HID,
                BS, HID, E, 1.f, 1, 0, 0, 0, 0, 0, 0);
        }
        // ---- FFN2: ff1 @ W2 + b2 -> P (f32) ----
        {
            const long H1E = (long)HID * E;
            f32_to_bf16_kernel<<<(int)((H1E + 255) / 256), 256, 0, stream>>>(W2 + l * H1E, wbuf, H1E);
            dim3 g(E / BN, BS / BM, 1);
            gemm_bf16_wmma<false, false, false><<<g, 256, 0, stream>>>(
                ff1b, HID, wbuf, E, P, E, b2 + l * E,
                BS, E, HID, 1.f, 1, 0, 0, 0, 0, 0, 0);
        }
        // ---- h = LN2(P + hB) -> hA (f32) ----
        ln_kernel<<<BS, 256, 0, stream>>>(P, hB, ln2g + l * E, ln2b + l * E, hA, nullptr);
        // ---- batchnorm over (batch, embed) per position; in-place hA + hb ----
        bn_kernel<<<Ss, 256, 0, stream>>>(hA, hb, bng + l * Ss, bnb + l * Ss);
    }

    // pooled mean over sequence, LM head, log-softmax
    pool_kernel<<<(Bb * E + 255) / 256, 256, 0, stream>>>(hA, pooled);
    {
        dim3 g((Vv + 255) / 256, Bb);
        lmhead_kernel<<<g, 256, 0, stream>>>(pooled, lmW, lmb, logits);
    }
    logsoftmax_kernel<<<Bb, 256, 0, stream>>>(logits, outp);
}
